// SparseMoE_38242388803770
// MI455X (gfx1250) — matrix operand order repaired and checked by reading, not compiled, not run on hardware
//
#include <hip/hip_runtime.h>
#include <math.h>

#define E_NUM   8
#define K_TOP   2
#define T_TOK   4096
#define D_DIM   512
#define H_DIM   2048
#define CAP     1075      // round(2*4096*1.05/8)
#define CAP_PAD 1088      // 34 tiles of 32
#define AROW    40        // padded LDS row pitch (ushorts) for A and Bt tiles

typedef __attribute__((ext_vector_type(16))) __bf16          v16bf;
typedef __attribute__((ext_vector_type(16))) unsigned short  v16u;
typedef __attribute__((ext_vector_type(8)))  unsigned short  v8u;
typedef __attribute__((ext_vector_type(4)))  unsigned short  v4u;
typedef __attribute__((ext_vector_type(8)))  float           v8f;

__device__ __forceinline__ unsigned short f2bf(float f) {
  unsigned u = __float_as_uint(f);
  u += 0x7FFFu + ((u >> 16) & 1u);          // round-to-nearest-even
  return (unsigned short)(u >> 16);
}
__device__ __forceinline__ float bf2f(unsigned short h) {
  return __uint_as_float(((unsigned)h) << 16);
}
// 16 contiguous ushorts -> bf16x16 fragment (two ds_load_b128)
__device__ __forceinline__ v16bf ldfrag(const unsigned short* p) {
  v8u a = *(const v8u*)(p);
  v8u b = *(const v8u*)(p + 8);
  v16u c = __builtin_shufflevector(a, b, 0,1,2,3,4,5,6,7,8,9,10,11,12,13,14,15);
  return __builtin_bit_cast(v16bf, c);
}

// ---------------- router: logits -> softmax -> stable top-2, imp[e] ----------------
__global__ __launch_bounds__(256) void moe_router(
    const float* __restrict__ x, const float* __restrict__ gw,
    int* __restrict__ topi, float* __restrict__ topg, float* __restrict__ imp)
{
  __shared__ float simp[E_NUM];
  const int tid = threadIdx.x, lane = tid & 31, wave = tid >> 5;
  if (tid < E_NUM) simp[tid] = 0.f;
  __syncthreads();

  const int token = blockIdx.x * 8 + wave;
  float acc[E_NUM];
#pragma unroll
  for (int e = 0; e < E_NUM; ++e) acc[e] = 0.f;

  const float* xr = x + (size_t)token * D_DIM + lane * 16;
#pragma unroll
  for (int i = 0; i < 4; ++i) {
    float4 xv = *(const float4*)(xr + i * 4);
#pragma unroll
    for (int e = 0; e < E_NUM; ++e) {
      float4 wv = *(const float4*)(gw + e * D_DIM + lane * 16 + i * 4);
      acc[e] += xv.x * wv.x + xv.y * wv.y + xv.z * wv.z + xv.w * wv.w;
    }
  }
#pragma unroll
  for (int e = 0; e < E_NUM; ++e)
#pragma unroll
    for (int off = 16; off > 0; off >>= 1)
      acc[e] += __shfl_xor(acc[e], off, 32);

  if (lane == 0) {
    float mx = acc[0];
    for (int e = 1; e < E_NUM; ++e) mx = fmaxf(mx, acc[e]);
    float g[E_NUM], s = 0.f;
    for (int e = 0; e < E_NUM; ++e) { g[e] = expf(acc[e] - mx); s += g[e]; }
    float inv = 1.f / s;
    for (int e = 0; e < E_NUM; ++e) { g[e] *= inv; atomicAdd(&simp[e], g[e]); }
    int i1 = 0;
    for (int e = 1; e < E_NUM; ++e) if (g[e] > g[i1]) i1 = e;   // strict > : first index wins ties
    int i2 = (i1 == 0) ? 1 : 0;
    for (int e = 0; e < E_NUM; ++e) if (e != i1 && g[e] > g[i2]) i2 = e;
    topi[token * 2 + 0] = i1; topi[token * 2 + 1] = i2;
    topg[token * 2 + 0] = g[i1]; topg[token * 2 + 1] = g[i2];
  }
  __syncthreads();
  if (tid < E_NUM) atomicAdd(&imp[tid], simp[tid]);
}

// -------- capacity scan (slot-major, matches reference cumsum order), builds lists --------
__global__ void moe_scan(const int* __restrict__ topi, const float* __restrict__ topg,
                         int* __restrict__ list, float* __restrict__ gateb,
                         int* __restrict__ counts)
{
  const int lane = threadIdx.x;       // single wave of 32
  __shared__ int base[E_NUM];
  if (lane < E_NUM) base[lane] = 0;
  __syncthreads();

  for (int s = 0; s < K_TOP; ++s) {
    for (int t0 = 0; t0 < T_TOK; t0 += 32) {
      const int t = t0 + lane;
      const int e = topi[t * K_TOP + s];
      const unsigned lt = (1u << lane) - 1u;
      unsigned mymask = 0u;
      int before = 0;
#pragma unroll
      for (int q = 0; q < E_NUM; ++q) {
        unsigned m = __builtin_amdgcn_ballot_w32(e == q);
        if (e == q)    before = __popc(m & lt);
        if (lane == q) mymask = m;
      }
      const int pos = base[e] + before;
      __syncthreads();
      if (lane < E_NUM) base[lane] += __popc(mymask);
      __syncthreads();
      if (pos < CAP) {
        list [e * CAP_PAD + pos] = t;
        gateb[e * CAP_PAD + pos] = topg[t * K_TOP + s];
      }
    }
  }
  if (lane < E_NUM) counts[lane] = min(base[lane], CAP);
}

// ---------------- aux losses ----------------
__global__ void moe_loss(const float* __restrict__ imp, const int* __restrict__ counts,
                         float* __restrict__ out)
{
  if (threadIdx.x == 0) {
    float m = 0.f;
    for (int e = 0; e < E_NUM; ++e) m += imp[e];
    m *= (1.f / E_NUM);
    float v = 0.f;
    for (int e = 0; e < E_NUM; ++e) { float d = imp[e] - m; v += d * d; }
    v *= (1.f / E_NUM);
    float r = sqrtf(v) / (m + 1e-6f);
    float imp_loss = r * r;

    float m2 = 0.f;
    for (int e = 0; e < E_NUM; ++e) m2 += (float)counts[e];
    m2 *= (1.f / E_NUM);
    float v2 = 0.f;
    for (int e = 0; e < E_NUM; ++e) { float d = (float)counts[e] - m2; v2 += d * d; }
    v2 *= (1.f / E_NUM);
    float r2 = sqrtf(v2) / (m2 + 1e-6f);
    float l_load = r2 * r2;

    out[(size_t)T_TOK * D_DIM + 0] = 0.5f * (imp_loss + l_load);
    out[(size_t)T_TOK * D_DIM + 1] = l_load;
  }
}

// ---------------- grouped GEMM with bf16x2 WMMA ----------------
// Block tile 32(M) x 256(N), k-steps of 32.  8 waves in 2x4 grid; wave tile 16x64.
// LDS tiles are stored in *fragment order*:
//   A  : [m][k] (contiguous K, as Bt)  -> lane frag = 16 contiguous ushorts
//   Bt : [n][k]                                 -> lane frag = 16 contiguous ushorts
// MODE 0: Hbuf[e,p,:] = gelu(x[tok(p),:] @ fc1_w[e] + fc1_b[e])           (KD=512,  ND=2048)
// MODE 1: out[tok(p),:] += gate(p) * (Hbuf[e,p,:] @ fc2_w[e] + fc2_b[e])  (KD=2048, ND=512)
template<int MODE, int KD, int ND>
__global__ __launch_bounds__(256) void moe_gemm(
    const float* __restrict__ Asrc,
    const float* __restrict__ Bw,
    const float* __restrict__ bias,
    const int*   __restrict__ list,
    const float* __restrict__ gateb,
    const int*   __restrict__ counts,
    float* __restrict__ Cdst)
{
  const int e   = blockIdx.z;
  const int m0  = blockIdx.y * 32;
  const int n0  = blockIdx.x * 256;
  const int cnt = counts[e];
  if (m0 >= cnt) return;                     // uniform early-out

  __shared__ __align__(16) unsigned short Ahi[32 * AROW],  Alo[32 * AROW];
  __shared__ __align__(16) unsigned short Bhi[256 * AROW], Blo[256 * AROW];
  __shared__ int   stok[32];
  __shared__ float sgate[32];

  const int tid  = threadIdx.x;
  const int lane = tid & 31, wave = tid >> 5;
  const int wm = wave >> 2, wn = wave & 3;   // 2x4 wave grid
  const int half = lane >> 4, l16 = lane & 15;

  if (tid < 32) {
    int p = m0 + tid;
    bool ok = p < cnt;
    stok[tid]  = ok ? list [e * CAP_PAD + p] : -1;
    sgate[tid] = ok ? gateb[e * CAP_PAD + p] : 0.f;
  }
  __syncthreads();

  // ---- per-thread A staging source (row gather) + contiguous-K dest slot ----
  const int arow = tid >> 3;                 // 0..31
  const int ac4  = (tid & 7) * 4;            // 0,4,..,28 (K group of 4)
  const int agrp = ac4 >> 3;                 // group-of-8 index
  const int agrp_p = agrp;  // [0,8,16,24] order (contiguous K, matches Bt)
  const int adst = agrp_p * 8 + (ac4 & 7);

  const float* asrc;
  if (MODE == 0) {
    int tk = stok[arow];
    asrc = (tk >= 0) ? (Asrc + (size_t)tk * KD + ac4) : (const float*)0;
  } else {
    int p = m0 + arow;
    asrc = (p < cnt) ? (Asrc + ((size_t)e * CAP_PAD + p) * KD + ac4) : (const float*)0;
  }

  v8f acc[4] = {v8f{}, v8f{}, v8f{}, v8f{}};

  for (int kb = 0; kb < KD; kb += 32) {
    __syncthreads();
    // ---- stage A (32x32), fp32 -> bf16 hi/lo, contiguous-K row-major ----
    {
      float4 v = make_float4(0.f, 0.f, 0.f, 0.f);
      if (asrc) v = *(const float4*)(asrc + kb);
      if (MODE == 0 && asrc && kb + 32 < KD) __builtin_prefetch(asrc + kb + 32, 0, 3);
      float vv[4] = {v.x, v.y, v.z, v.w};
      v4u h4, l4;
#pragma unroll
      for (int j = 0; j < 4; ++j) {
        unsigned short h = f2bf(vv[j]);
        h4[j] = h;
        l4[j] = f2bf(vv[j] - bf2f(h));
      }
      *(v4u*)&Ahi[arow * AROW + adst] = h4;
      *(v4u*)&Alo[arow * AROW + adst] = l4;
    }
    // ---- stage B (32K x 256N) transposed into Bt[n][k] via 4x4 micro-transpose ----
#pragma unroll
    for (int pass = 0; pass < 2; ++pass) {
      const int mt = pass * 256 + tid;       // 512 micro-tiles of 4x4
      const int kq = mt >> 6, nq = mt & 63;
      const float* bp = Bw + ((size_t)e * KD + kb + kq * 4) * ND + n0 + nq * 4;
      float4 r0 = *(const float4*)(bp);
      float4 r1 = *(const float4*)(bp + ND);
      float4 r2 = *(const float4*)(bp + 2 * (size_t)ND);
      float4 r3 = *(const float4*)(bp + 3 * (size_t)ND);
      if (kb + 32 < KD) __builtin_prefetch(bp + 32 * (size_t)ND, 0, 3);
      float cols[4][4] = {{r0.x, r1.x, r2.x, r3.x}, {r0.y, r1.y, r2.y, r3.y},
                          {r0.z, r1.z, r2.z, r3.z}, {r0.w, r1.w, r2.w, r3.w}};
#pragma unroll
      for (int j = 0; j < 4; ++j) {
        v4u h4, l4;
#pragma unroll
        for (int r = 0; r < 4; ++r) {
          unsigned short h = f2bf(cols[j][r]);
          h4[r] = h;
          l4[r] = f2bf(cols[j][r] - bf2f(h));
        }
        *(v4u*)&Bhi[(nq * 4 + j) * AROW + kq * 4] = h4;
        *(v4u*)&Blo[(nq * 4 + j) * AROW + kq * 4] = l4;
      }
    }
    __syncthreads();

    // ---- fragments: contiguous 32B per lane -> 2x ds_load_b128 each ----
    const unsigned short* ap = &Ahi[(wm * 16 + l16) * AROW + half * 16];
    const unsigned short* lp = &Alo[(wm * 16 + l16) * AROW + half * 16];
    v16bf ah  = ldfrag(ap);
    v16bf alv = ldfrag(lp);
#pragma unroll
    for (int sub = 0; sub < 4; ++sub) {
      const int br = (wn * 64 + sub * 16 + l16) * AROW + half * 16;
      v16bf bh = ldfrag(&Bhi[br]);
      v16bf bl = ldfrag(&Blo[br]);
      // bf16x2: acc += al*bh + ah*bl + ah*bh (f32 accumulate)
      acc[sub] = __builtin_amdgcn_wmma_f32_16x16x32_bf16(false, alv, false, bh, (short)0, acc[sub], false, false);
      acc[sub] = __builtin_amdgcn_wmma_f32_16x16x32_bf16(false, ah,  false, bl, (short)0, acc[sub], false, false);
      acc[sub] = __builtin_amdgcn_wmma_f32_16x16x32_bf16(false, ah,  false, bh, (short)0, acc[sub], false, false);
    }
  }

  // ---- epilogue (C/D layout: VGPR r -> row 8*(lane/16)+r, col = lane%16) ----
#pragma unroll
  for (int r = 0; r < 8; ++r) {
    const int lr = wm * 16 + half * 8 + r;
    const int p  = m0 + lr;
    if (p >= cnt) continue;
#pragma unroll
    for (int sub = 0; sub < 4; ++sub) {
      const int n = n0 + wn * 64 + sub * 16 + l16;
      float v = acc[sub][r] + bias[e * ND + n];
      if (MODE == 0) {
        v = 0.5f * v * (1.f + erff(v * 0.70710678118f));  // exact GELU
        Cdst[((size_t)e * CAP_PAD + p) * ND + n] = v;
      } else {
        atomicAdd(&Cdst[(size_t)stok[lr] * ND + n], sgate[lr] * v);
      }
    }
  }
}

extern "C" void kernel_launch(void* const* d_in, const int* in_sizes, int n_in,
                              void* d_out, int out_size, void* d_ws, size_t ws_size,
                              hipStream_t stream) {
  const float* x    = (const float*)d_in[0];
  const float* gw   = (const float*)d_in[1];
  const float* fc1w = (const float*)d_in[2];
  const float* fc1b = (const float*)d_in[3];
  const float* fc2w = (const float*)d_in[4];
  const float* fc2b = (const float*)d_in[5];
  float* out = (float*)d_out;

  char* wsb = (char*)d_ws;
  size_t off = 0;
  auto take = [&](size_t bytes) -> char* {
    char* p = wsb + off;
    off = (off + bytes + 255) & ~(size_t)255;
    return p;
  };
  int*   topi   = (int*)  take((size_t)T_TOK * K_TOP * sizeof(int));
  float* topg   = (float*)take((size_t)T_TOK * K_TOP * sizeof(float));
  float* imp    = (float*)take(E_NUM * sizeof(float));
  int*   counts = (int*)  take(E_NUM * sizeof(int));
  int*   list   = (int*)  take((size_t)E_NUM * CAP_PAD * sizeof(int));
  float* gateb  = (float*)take((size_t)E_NUM * CAP_PAD * sizeof(float));
  float* Hbuf   = (float*)take((size_t)E_NUM * CAP_PAD * H_DIM * sizeof(float));

  // out is accumulated via atomics -> zero every call (graph replays don't re-poison)
  hipMemsetAsync(d_out, 0, (size_t)out_size * sizeof(float), stream);
  hipMemsetAsync(imp, 0, E_NUM * sizeof(float), stream);

  moe_router<<<T_TOK / 8, 256, 0, stream>>>(x, gw, topi, topg, imp);
  moe_scan  <<<1, 32, 0, stream>>>(topi, topg, list, gateb, counts);
  moe_loss  <<<1, 32, 0, stream>>>(imp, counts, out);
  moe_gemm<0, D_DIM, H_DIM><<<dim3(H_DIM / 256, CAP_PAD / 32, E_NUM), 256, 0, stream>>>(
      x, fc1w, fc1b, list, gateb, counts, Hbuf);
  moe_gemm<1, H_DIM, D_DIM><<<dim3(D_DIM / 256, CAP_PAD / 32, E_NUM), 256, 0, stream>>>(
      Hbuf, fc2w, fc2b, list, gateb, counts, out);
}
